// MPNN_38878043963729
// MI455X (gfx1250) — matrix operand order repaired and checked by reading, not compiled, not run on hardware
//
#include <hip/hip_runtime.h>

typedef __attribute__((ext_vector_type(16))) __bf16   v16bf;
typedef __attribute__((ext_vector_type(8)))  float    v8f;
typedef __attribute__((ext_vector_type(8)))  unsigned v8u;
typedef __attribute__((ext_vector_type(4)))  unsigned u32x4;
typedef __attribute__((ext_vector_type(4)))  unsigned v4u;
typedef __attribute__((ext_vector_type(8)))  int      v8i;
typedef __attribute__((ext_vector_type(4)))  int      v4i;

#define Bb   8
#define Nn   256
#define Hh   128
#define MIDd 128
#define FINf 256

// LDS row strides (dwords), padded so lane r's b128 frag read lands on bank quad 4r
#define EPITCH 132   // f32 tile rows: 128 data + 4 pad dwords (TDM pad feature)
#define BPITCH 68    // bf16 tile rows: 64 data dwords + 4 pad
#define FPITCH 132   // k_node feat rows: 128 data dwords (256 bf16) + 4 pad

// ---------- helpers ----------
static __device__ __forceinline__ v8f wmma_bf16f32(v8u a, v8u b, v8f c) {
  return __builtin_amdgcn_wmma_f32_16x16x32_bf16(
      false, __builtin_bit_cast(v16bf, a),
      false, __builtin_bit_cast(v16bf, b),
      (short)0, c, false, false);
}

// split-f32 -> bf16 hi (truncate) and bf16 lo (residual), packed pairs
static __device__ __forceinline__ unsigned pack_hi2(float a, float b) {
  return (__float_as_uint(a) >> 16) | (__float_as_uint(b) & 0xFFFF0000u);
}
static __device__ __forceinline__ unsigned pack_lo2(float a, float b) {
  float ra = a - __uint_as_float(__float_as_uint(a) & 0xFFFF0000u);
  float rb = b - __uint_as_float(__float_as_uint(b) & 0xFFFF0000u);
  return (__float_as_uint(ra) >> 16) | (__float_as_uint(rb) & 0xFFFF0000u);
}

static __device__ __forceinline__ v8u ld8(const unsigned* p) {
  u32x4 a = *(const u32x4*)p;
  u32x4 b = *(const u32x4*)(p + 4);
  return __builtin_shufflevector(a, b, 0, 1, 2, 3, 4, 5, 6, 7);
}

// TDM: load e_features[b, it*16 .. +15, j, 0:128] (f32) into LDS at ldsOff,
// with 4-dword pad after each 128-dword row (-> row stride EPITCH dwords).
static __device__ __forceinline__ void tdm_load_tile(const float* ef, int b, int j,
                                                     int it, unsigned ldsOff) {
  unsigned long long ga =
      (unsigned long long)(const void*)(ef + ((((size_t)b * Nn + (size_t)it * 16) * Nn + j) * Hh));
  v4u g0 = { 1u,                              // count=1 valid descriptor
             ldsOff,                          // lds_addr (bytes)
             (unsigned)ga,                    // global_addr[31:0]
             (unsigned)(ga >> 32) | 0x80000000u };  // global_addr[56:32] | type=2<<30
  v8i g1 = { (int)0x07920000,  // data_size=4B(2<<16), pad_en(1<<20), interval=128dw(6<<22), amount=4dw(3<<25)
             (int)(128u << 16),   // abar=0 | tensor_dim0[15:0]=128
             (int)(256u << 16),   // tensor_dim0[31:16]=0 | tensor_dim1[15:0]=256
             (int)(128u << 16),   // tensor_dim1[31:16]=0 | tile_dim0=128
             16,                  // tile_dim1=16 | tile_dim2=0
             32768,               // tensor_dim0_stride = N*HE (data_size units)
             0, 0 };              // dim1 stride unused (2-D tile)
  v4i gz4 = { 0, 0, 0, 0 };
  v8i gz8 = { 0, 0, 0, 0, 0, 0, 0, 0 };
  __builtin_amdgcn_tensor_load_to_lds(g0, g1, gz4, gz4, gz8, 0);
}

// ---------- kernel 0: msg_g = g @ Wg + bg  [8,128] ----------
__global__ __launch_bounds__(128) void k_msgg(const float* __restrict__ g,
                                              const float* __restrict__ Wg,
                                              const float* __restrict__ bg,
                                              float* __restrict__ msg_g) {
  int b = blockIdx.x, m = threadIdx.x;
  float acc = bg[m];
#pragma unroll 8
  for (int k = 0; k < Hh; ++k) acc += g[b * Hh + k] * Wg[k * MIDd + m];
  msg_g[b * MIDd + m] = acc;
}

// ---------- kernel 1: pack weights into WMMA B-fragment layout (bf16 hi/lo) ----------
// B-frag (32x16 bf16): lane l holds N = nt*16 + (l&15); VGPR r pair p -> K = kk + 16*(l>>4) + 2r + p.
__global__ __launch_bounds__(32) void k_prep(const float* __restrict__ W1,
                                             const float* __restrict__ W2,
                                             const float* __restrict__ Wo1,
                                             const float* __restrict__ We,
                                             unsigned* __restrict__ nodeHi,
                                             unsigned* __restrict__ nodeLo,
                                             unsigned* __restrict__ eHi,
                                             unsigned* __restrict__ eLo) {
  int f = blockIdx.x;
  int l = threadIdx.x;
  if (f < 8 * 24) {  // [W1|W2|Wo1]: K=256 (8 kk-tiles), 384 cols (24 n-tiles)
    int kk8 = f / 24, nt = f % 24;
    int kk = kk8 * 32;
    int n = nt * 16 + (l & 15);
    const float* src;
    int col;
    if (n < 128)      { src = W1;  col = n; }
    else if (n < 256) { src = W2;  col = n - 128; }
    else              { src = Wo1; col = n - 256; }
    unsigned* dh = nodeHi + f * 256 + l * 8;
    unsigned* dl = nodeLo + f * 256 + l * 8;
#pragma unroll
    for (int r = 0; r < 8; ++r) {
      int k0 = kk + ((l >> 4) << 4) + 2 * r;
      float a = src[k0 * MIDd + col];
      float b = src[(k0 + 1) * MIDd + col];
      dh[r] = pack_hi2(a, b);
      dl[r] = pack_lo2(a, b);
    }
  } else {           // We: K=128 (4 kk-tiles), 128 cols (8 n-tiles)
    int f2 = f - 8 * 24;
    int kk4 = f2 / 8, nt = f2 % 8;
    int kk = kk4 * 32;
    int n = nt * 16 + (l & 15);
    unsigned* dh = eHi + f2 * 256 + l * 8;
    unsigned* dl = eLo + f2 * 256 + l * 8;
#pragma unroll
    for (int r = 0; r < 8; ++r) {
      int k0 = kk + ((l >> 4) << 4) + 2 * r;
      float a = We[k0 * MIDd + n];
      float b = We[(k0 + 1) * MIDd + n];
      dh[r] = pack_hi2(a, b);
      dl[r] = pack_lo2(a, b);
    }
  }
}

// ---------- kernel 2: node GEMM: [msg1 | msg2+msg_g | feat@Wo1+bo1] ----------
__global__ __launch_bounds__(128) void k_node(const float* __restrict__ nf,
                                              const float* __restrict__ hid,
                                              const float* __restrict__ b1,
                                              const float* __restrict__ b2,
                                              const float* __restrict__ bo1,
                                              const float* __restrict__ msg_g,
                                              const unsigned* __restrict__ nodeHi,
                                              const unsigned* __restrict__ nodeLo,
                                              float* __restrict__ msg1,
                                              float* __restrict__ msg2p,
                                              float* __restrict__ basep) {
  __shared__ __align__(16) unsigned fhi[16 * FPITCH];  // 16 rows x 256 bf16 (+pad)
  __shared__ __align__(16) unsigned flo[16 * FPITCH];
  int t = threadIdx.x;
  int nodeBase = blockIdx.x * 16;  // global node = b*256 + n
  {
    int row = t >> 3;
    int c0 = (t & 7) * 32;  // 32 floats, entirely within one source half
    int node = nodeBase + row;
    const float* s = (c0 < 128) ? (nf + node * Hh + c0) : (hid + node * Hh + (c0 - 128));
    unsigned* dh = &fhi[row * FPITCH + (c0 >> 1)];
    unsigned* dl = &flo[row * FPITCH + (c0 >> 1)];
#pragma unroll
    for (int q = 0; q < 16; ++q) {
      float a = s[2 * q], b = s[2 * q + 1];
      dh[q] = pack_hi2(a, b);
      dl[q] = pack_lo2(a, b);
    }
  }
  __syncthreads();

  int l = t & 31, wid = t >> 5;
  int half = l >> 4;
  int row = l & 15;
  for (int ni = 0; ni < 6; ++ni) {
    int nt = wid * 6 + ni;
    v8f c = {};
#pragma unroll
    for (int kk8 = 0; kk8 < 8; ++kk8) {
      int adw = row * FPITCH + kk8 * 16 + 8 * half;
      v8u ah = ld8(&fhi[adw]);
      v8u al = ld8(&flo[adw]);
      v8u bhv = ld8(nodeHi + (kk8 * 24 + nt) * 256 + l * 8);
      v8u blv = ld8(nodeLo + (kk8 * 24 + nt) * 256 + l * 8);
      c = wmma_bf16f32(ah, bhv, c);
      c = wmma_bf16f32(al, bhv, c);
      c = wmma_bf16f32(ah, blv, c);
    }
    int col = nt * 16 + row;  // 0..383
    int b = nodeBase >> 8;
    float bias;
    float* dst;
    int dc;
    if (col < 128)      { bias = b1[col];                                     dst = msg1;  dc = col; }
    else if (col < 256) { bias = b2[col - 128] + msg_g[b * MIDd + col - 128]; dst = msg2p; dc = col - 128; }
    else                { bias = bo1[col - 256];                              dst = basep; dc = col - 256; }
#pragma unroll
    for (int r = 0; r < 8; ++r) {
      int node = nodeBase + r + 8 * half;
      dst[node * MIDd + dc] = c[r] + bias;
    }
  }
}

// ---------- kernel 3: TDM-prefetched edge GEMM + masked max + output GEMM ----------
__global__ __launch_bounds__(128) void k_edge(const float* __restrict__ ef,
                                              const float* __restrict__ adj,
                                              const float* __restrict__ msg1,
                                              const float* __restrict__ msg2p,
                                              const float* __restrict__ be,
                                              const float* __restrict__ basep,
                                              const float* __restrict__ Wo2,
                                              const float* __restrict__ bo2,
                                              const unsigned* __restrict__ eHi,
                                              const unsigned* __restrict__ eLo,
                                              float* __restrict__ out) {
  __shared__ __align__(16) float eBuf[2][16 * EPITCH];  // TDM landing pads (f32, padded rows)
  __shared__ __align__(16) unsigned bHi[16 * BPITCH];   // bf16 hi tile
  __shared__ __align__(16) unsigned bLo[16 * BPITCH];   // bf16 lo tile
  __shared__ float adjs[16];
  __shared__ float msgs[128];
  int t = threadIdx.x;
  int b = blockIdx.x >> 8, j = blockIdx.x & 255;
  int l = t & 31, wid = t >> 5;
  int half = l >> 4;
  int arow = l & 15;

  // kick off TDM for tile 0
  if (wid == 0) tdm_load_tile(ef, b, j, 0, (unsigned)(unsigned long long)(uintptr_t)&eBuf[0][0]);

  // preload We B-fragments into registers: 2 n-tiles x 4 k-tiles x hi/lo
  int nt0 = wid * 2;
  v8u Bh[2][4], Bl[2][4];
#pragma unroll
  for (int ni = 0; ni < 2; ++ni)
#pragma unroll
    for (int kk4 = 0; kk4 < 4; ++kk4) {
      Bh[ni][kk4] = ld8(eHi + (kk4 * 8 + nt0 + ni) * 256 + l * 8);
      Bl[ni][kk4] = ld8(eLo + (kk4 * 8 + nt0 + ni) * 256 + l * 8);
    }

  int m0 = nt0 * 16 + arow;
  int m1 = m0 + 16;
  int nj = b * Nn + j;
  float s1[2] = { msg1[nj * MIDd + m0] + be[m0], msg1[nj * MIDd + m1] + be[m1] };
  float mx[2] = { -__builtin_inff(), -__builtin_inff() };

  for (int it = 0; it < 16; ++it) {
    if (wid == 0) {
      if (it + 1 < 16) {  // prefetch next tile into the other buffer
        tdm_load_tile(ef, b, j, it + 1,
                      (unsigned)(unsigned long long)(uintptr_t)&eBuf[(it + 1) & 1][0]);
        __builtin_amdgcn_s_wait_tensorcnt(1);  // tile `it` landed, `it+1` in flight
      } else {
        __builtin_amdgcn_s_wait_tensorcnt(0);  // last tile landed
      }
    }
    __syncthreads();  // eBuf[it&1] visible to all; previous bHi/bLo reads done

    {  // convert landed f32 tile -> bf16 hi/lo (bank-conflict-padded)
      int row = t >> 3, seg = t & 7;
      const float* s = &eBuf[it & 1][row * EPITCH + seg * 16];
      unsigned* dh = &bHi[row * BPITCH + seg * 8];
      unsigned* dl = &bLo[row * BPITCH + seg * 8];
#pragma unroll
      for (int q = 0; q < 8; ++q) {
        float a = s[2 * q], bb = s[2 * q + 1];
        dh[q] = pack_hi2(a, bb);
        dl[q] = pack_lo2(a, bb);
      }
      if (t < 16) adjs[t] = adj[(b * Nn + (it * 16 + t)) * Nn + j];
    }
    __syncthreads();

#pragma unroll
    for (int ni = 0; ni < 2; ++ni) {
      int m = ni ? m1 : m0;
      v8f c;
#pragma unroll
      for (int r = 0; r < 8; ++r)  // C init = msg2' tile (free add, L2-hot)
        c[r] = msg2p[(b * Nn + it * 16 + r + 8 * half) * MIDd + m];
#pragma unroll
      for (int kk4 = 0; kk4 < 4; ++kk4) {
        int adw = arow * BPITCH + kk4 * 16 + 8 * half;
        v8u ah = ld8(&bHi[adw]);
        v8u al = ld8(&bLo[adw]);
        c = wmma_bf16f32(ah, Bh[ni][kk4], c);
        c = wmma_bf16f32(al, Bh[ni][kk4], c);
        c = wmma_bf16f32(ah, Bl[ni][kk4], c);
      }
      float s = s1[ni];
      float m2 = mx[ni];
#pragma unroll
      for (int r = 0; r < 8; ++r) {
        float v = adjs[r + 8 * half] * (c[r] + s);
        m2 = fmaxf(m2, v);
      }
      mx[ni] = m2;
    }
  }
  // cross-lane max over the two row-halves (lane ^ 16 holds the same column)
  mx[0] = fmaxf(mx[0], __shfl_xor(mx[0], 16, 32));
  mx[1] = fmaxf(mx[1], __shfl_xor(mx[1], 16, 32));
  if (l < 16) { msgs[m0] = mx[0]; msgs[m1] = mx[1]; }
  __syncthreads();

  // fused output: out[b,j,t] = base + bo2[t] + msgs @ Wo2[:,t]
  float acc = bo2[t] + basep[nj * MIDd + t];
#pragma unroll 8
  for (int m = 0; m < MIDd; ++m) acc += msgs[m] * Wo2[m * MIDd + t];
  out[nj * MIDd + t] = acc;
}

// ---------- launch ----------
extern "C" void kernel_launch(void* const* d_in, const int* in_sizes, int n_in,
                              void* d_out, int out_size, void* d_ws, size_t ws_size,
                              hipStream_t stream) {
  (void)in_sizes; (void)n_in; (void)out_size; (void)ws_size;
  const float* hidden = (const float*)d_in[0];
  const float* nfeat  = (const float*)d_in[1];
  const float* efeat  = (const float*)d_in[2];
  const float* gfeat  = (const float*)d_in[3];
  const float* adj    = (const float*)d_in[4];
  const float* W1  = (const float*)d_in[5];
  const float* b1  = (const float*)d_in[6];
  const float* W2  = (const float*)d_in[7];
  const float* b2  = (const float*)d_in[8];
  const float* We  = (const float*)d_in[9];
  const float* be  = (const float*)d_in[10];
  const float* Wg  = (const float*)d_in[11];
  const float* bg  = (const float*)d_in[12];
  const float* Wo1 = (const float*)d_in[13];
  const float* bo1 = (const float*)d_in[14];
  const float* Wo2 = (const float*)d_in[15];
  const float* bo2 = (const float*)d_in[16];
  float* out = (float*)d_out;

  char* ws = (char*)d_ws;
  size_t off = 0;
  float* msg_g = (float*)(ws + off); off += (size_t)Bb * MIDd * 4;
  float* msg1  = (float*)(ws + off); off += (size_t)Bb * Nn * MIDd * 4;
  float* msg2p = (float*)(ws + off); off += (size_t)Bb * Nn * MIDd * 4;
  float* basep = (float*)(ws + off); off += (size_t)Bb * Nn * MIDd * 4;
  unsigned* nodeHi = (unsigned*)(ws + off); off += (size_t)8 * 24 * 256 * 4;
  unsigned* nodeLo = (unsigned*)(ws + off); off += (size_t)8 * 24 * 256 * 4;
  unsigned* eHi    = (unsigned*)(ws + off); off += (size_t)4 * 8 * 256 * 4;
  unsigned* eLo    = (unsigned*)(ws + off); off += (size_t)4 * 8 * 256 * 4;

  k_msgg<<<Bb, 128, 0, stream>>>(gfeat, Wg, bg, msg_g);
  k_prep<<<8 * 24 + 4 * 8, 32, 0, stream>>>(W1, W2, Wo1, We, nodeHi, nodeLo, eHi, eLo);
  k_node<<<(Bb * Nn) / 16, 128, 0, stream>>>(nfeat, hidden, b1, b2, bo1, msg_g,
                                             nodeHi, nodeLo, msg1, msg2p, basep);
  k_edge<<<Bb * Nn, 128, 0, stream>>>(efeat, adj, msg1, msg2p, be, basep, Wo2, bo2,
                                      eHi, eLo, out);
}
